// Attention_40570261078258
// MI455X (gfx1250) — matrix operand order repaired and checked by reading, not compiled, not run on hardware
//
#include <hip/hip_runtime.h>
#include <hip/hip_bf16.h>

#define N_  8
#define C_  128
#define L_  4096
#define LT_ 256   // L/16  (16-row tiles)
#define MC_ 128   // L/32  (32-col chunks)

typedef __attribute__((ext_vector_type(16))) __bf16       v16bf;
typedef __attribute__((ext_vector_type(8)))  float        v8f;
typedef __attribute__((ext_vector_type(4)))  unsigned int v4u;

union Frag { v16bf v; v4u q[2]; unsigned short u[16]; };

__device__ __forceinline__ unsigned short f2bf(float f) {
    unsigned u = __float_as_uint(f);
    return (unsigned short)((u + 0x8000u) >> 16);   // round-half-up bf16
}

// ---------------------------------------------------------------------------
// Kernel 0: convert Wq/Wk/Wv (f32, [o][c] row-major) into bf16 B-fragment
// layout for D[l,o] = sum_c A[l,c]*B[c,o], B[c,o] = W[o,c].
// B-frag lane t: col o = ot*16 + (t%16), elems i: c = kc*32 + (t/16)*16 + i.
// ---------------------------------------------------------------------------
__global__ void wswizzle_kernel(const float* __restrict__ Wq,
                                const float* __restrict__ Wk,
                                const float* __restrict__ Wv,
                                __bf16* __restrict__ WbOut) {
    int gid = blockIdx.x * blockDim.x + threadIdx.x;    // mat*1024 + ot*128 + kc*32 + t
    if (gid >= 3 * 8 * 4 * 32) return;
    int t   = gid & 31;
    int kc  = (gid >> 5) & 3;
    int ot  = (gid >> 7) & 7;
    int mat = gid >> 10;
    const float* Wsrc = (mat == 0) ? Wq : (mat == 1) ? Wk : Wv;
    int o     = ot * 16 + (t & 15);
    int cbase = kc * 32 + (t >> 4) * 16;
    Frag f;
#pragma unroll
    for (int i = 0; i < 16; ++i) f.u[i] = f2bf(Wsrc[o * C_ + cbase + i]);
    ((v16bf*)WbOut)[gid] = f.v;
}

// ---------------------------------------------------------------------------
// Kernel 1: QKV projection. One wave per (n, 16-row l-tile).
//   D[l,o] = bias[o] + sum_c X[c,l] * W[o,c]   via v_wmma_f32_16x16x32_bf16
// Outputs written pre-swizzled:
//   Qa: A-fragment layout      (for S = Q @ K^T)
//   Kb: B-fragment layout      (B[c,m] = K[m,c])
//   Vb: B-fragment layout      (B[m,c] = V[m,c], 32-m chunks)
// ---------------------------------------------------------------------------
__global__ void __launch_bounds__(32) qkv_kernel(
    const float* __restrict__ x,
    const float* __restrict__ bq, const float* __restrict__ bk,
    const float* __restrict__ bv,
    const __bf16* __restrict__ Wb,
    __bf16* __restrict__ Qa, __bf16* __restrict__ Kb, __bf16* __restrict__ Vb)
{
    __shared__ __align__(16) unsigned short tile[16 * C_];
    int t  = threadIdx.x;
    int n  = blockIdx.x >> 8;
    int lt = blockIdx.x & 255;
    int tl = t & 15, th = t >> 4;

    // A-frags of X^T: lane t row l = lt*16 + tl; elems i -> c = kc*32 + th*16 + i
    // (contiguous k order, matching Wb B-frags)
    Frag xfrag[4];
    int lrow = lt * 16 + tl;
#pragma unroll
    for (int kc = 0; kc < 4; ++kc) {
#pragma unroll
        for (int i = 0; i < 16; ++i) {
            int c = kc * 32 + th * 16 + i;
            xfrag[kc].u[i] = f2bf(x[((size_t)n * C_ + c) * L_ + lrow]);
        }
    }

    const v16bf* WbV = (const v16bf*)Wb;
    for (int mat = 0; mat < 3; ++mat) {
        const float* bias = (mat == 0) ? bq : (mat == 1) ? bk : bv;
#pragma unroll
        for (int ot = 0; ot < 8; ++ot) {
            float bb = bias[ot * 16 + tl];        // D col depends only on lane%16
            v8f D;
#pragma unroll
            for (int j = 0; j < 8; ++j) D[j] = bb;
#pragma unroll
            for (int kc = 0; kc < 4; ++kc) {
                Frag wf; wf.v = WbV[((mat * 8 + ot) * 4 + kc) * 32 + t];
                D = __builtin_amdgcn_wmma_f32_16x16x32_bf16(
                        false, xfrag[kc].v, false, wf.v, (short)0, D, false, false);
            }
            // C/D layout: VGPR j -> row j + 8*(lane/16), col = ot*16 + lane%16
#pragma unroll
            for (int j = 0; j < 8; ++j)
                tile[(j + 8 * th) * C_ + ot * 16 + tl] = f2bf(D[j]);
        }
        __syncthreads();

        if (mat == 0) {          // Q -> A-fragment layout
            v16bf* QaV = (v16bf*)Qa;
#pragma unroll
            for (int kc = 0; kc < 4; ++kc) {
                Frag f;
                f.q[0] = *(const v4u*)&tile[tl * C_ + kc * 32 + th * 16];
                f.q[1] = *(const v4u*)&tile[tl * C_ + kc * 32 + th * 16 + 8];
                QaV[(((size_t)n * LT_ + lt) * 4 + kc) * 32 + t] = f.v;
            }
        } else if (mat == 1) {   // K -> B-fragment layout (cols = m rows of K)
            v16bf* KbV = (v16bf*)Kb;
#pragma unroll
            for (int kc = 0; kc < 4; ++kc) {
                Frag f;
                const v4u* p = (const v4u*)&tile[tl * C_ + kc * 32 + th * 16];
                f.q[0] = p[0]; f.q[1] = p[1];
                KbV[(((size_t)n * LT_ + lt) * 4 + kc) * 32 + t] = f.v;
            }
        } else {                 // V -> B-fragment layout over 32-m chunks
            v16bf* VbV = (v16bf*)Vb;
            int mcIdx = lt >> 1, g = lt & 1;     // this tile is half of chunk mcIdx
#pragma unroll
            for (int rep = 0; rep < 4; ++rep) {
                int ct = rep * 2 + th;
                Frag f;
#pragma unroll
                for (int i = 0; i < 16; ++i)
                    f.u[i] = tile[i * C_ + ct * 16 + tl];
                VbV[(((size_t)n * MC_ + mcIdx) * 8 + ct) * 32 + (g * 16 + tl)] = f.v;
            }
        }
        __syncthreads();
    }
}

// ---------------------------------------------------------------------------
// Kernel 2: fused scores -> batch-axis softmax -> attn@V -> residual add.
// Block = one 16-row l-tile, wave w = batch n=w. Softmax couples the 8 waves
// elementwise over n only, handled via LDS e-buffer + cross-wave sum.
// ---------------------------------------------------------------------------
__global__ void __launch_bounds__(256) attn_kernel(
    const float* __restrict__ x,
    const __bf16* __restrict__ Qa, const __bf16* __restrict__ Kb,
    const __bf16* __restrict__ Vb, float* __restrict__ y)
{
    __shared__ __align__(16) float eBuf[N_][512];   // exp(scores), per batch
    __shared__ __align__(16) float sumBuf[512];     // sum over n

    int tid = threadIdx.x;
    int n = tid >> 5;
    int t = tid & 31;
    int tl = t & 15, th = t >> 4;
    int lt = blockIdx.x;

    const v16bf* QaV = (const v16bf*)Qa;
    const v16bf* KbV = (const v16bf*)Kb;
    const v16bf* VbV = (const v16bf*)Vb;

    Frag qfrag[4];
#pragma unroll
    for (int kc = 0; kc < 4; ++kc)
        qfrag[kc].v = QaV[(((size_t)n * LT_ + lt) * 4 + kc) * 32 + t];

    v8f outAcc[8];
#pragma unroll
    for (int ct = 0; ct < 8; ++ct)
#pragma unroll
        for (int j = 0; j < 8; ++j) outAcc[ct][j] = 0.f;

    const float SC = 0.12753139f;   // log2(e) / sqrt(128)

    for (int mc = 0; mc < MC_; ++mc) {
        if (mc + 1 < MC_)   // prefetch next K chunk (L2-resident hint)
            __builtin_prefetch(&KbV[(((size_t)n * LT_ + (mc + 1) * 2) * 4) * 32 + t], 0, 0);

        // ---- S = Q @ K^T for two 16-col tiles, then e = exp(S/sqrt(C))
#pragma unroll
        for (int u = 0; u < 2; ++u) {
            int mt = mc * 2 + u;
            v8f S;
#pragma unroll
            for (int j = 0; j < 8; ++j) S[j] = 0.f;
#pragma unroll
            for (int kc = 0; kc < 4; ++kc) {
                Frag kf; kf.v = KbV[(((size_t)n * LT_ + mt) * 4 + kc) * 32 + t];
                S = __builtin_amdgcn_wmma_f32_16x16x32_bf16(
                        false, qfrag[kc].v, false, kf.v, (short)0, S, false, false);
            }
#pragma unroll
            for (int j = 0; j < 8; ++j)
                eBuf[n][(j + 8 * th) * 32 + u * 16 + tl] = exp2f(S[j] * SC);
        }
        __syncthreads();

        // ---- softmax denominator: sum over the 8 batches, split across waves
        {
            int q0 = n * 64 + t * 2;
            float s0 = 0.f, s1 = 0.f;
#pragma unroll
            for (int nn = 0; nn < N_; ++nn) { s0 += eBuf[nn][q0]; s1 += eBuf[nn][q0 + 1]; }
            sumBuf[q0] = s0; sumBuf[q0 + 1] = s1;
        }
        __syncthreads();

        // ---- attn = e / sum, repacked into A-fragment layout (LDS = transpose)
        Frag af;
#pragma unroll
        for (int h = 0; h < 2; ++h) {
            int base = tl * 32 + th * 16 + h * 8;
#pragma unroll
            for (int i = 0; i < 8; ++i) {
                float e = eBuf[n][base + i];
                float s = sumBuf[base + i];
                af.u[h * 8 + i] = f2bf(e * __builtin_amdgcn_rcpf(s));
            }
        }

        // ---- out += attn @ V
#pragma unroll
        for (int ct = 0; ct < 8; ++ct) {
            Frag vf; vf.v = VbV[(((size_t)n * MC_ + mc) * 8 + ct) * 32 + t];
            outAcc[ct] = __builtin_amdgcn_wmma_f32_16x16x32_bf16(
                             false, af.v, false, vf.v, (short)0, outAcc[ct], false, false);
        }
    }

    // ---- y = x + out; (N,L,C) flat == (N,C,H,W) flat (faithful reshape)
#pragma unroll
    for (int ct = 0; ct < 8; ++ct) {
#pragma unroll
        for (int j = 0; j < 8; ++j) {
            int l = lt * 16 + j + 8 * th;
            int c = ct * 16 + tl;
            size_t idx = (size_t)n * (size_t)(L_ * C_) + (size_t)l * C_ + c;
            y[idx] = x[idx] + outAcc[ct][j];
        }
    }
}

// ---------------------------------------------------------------------------
extern "C" void kernel_launch(void* const* d_in, const int* in_sizes, int n_in,
                              void* d_out, int out_size, void* d_ws, size_t ws_size,
                              hipStream_t stream)
{
    const float* x  = (const float*)d_in[0];
    const float* Wq = (const float*)d_in[1];
    const float* bq = (const float*)d_in[2];
    const float* Wk = (const float*)d_in[3];
    const float* bk = (const float*)d_in[4];
    const float* Wv = (const float*)d_in[5];
    const float* bv = (const float*)d_in[6];
    float* y = (float*)d_out;

    char* ws = (char*)d_ws;
    __bf16* Qa = (__bf16*)(ws);                               // 8 MiB
    __bf16* Kb = (__bf16*)(ws + (size_t)8  * 1024 * 1024);    // 8 MiB
    __bf16* Vb = (__bf16*)(ws + (size_t)16 * 1024 * 1024);    // 8 MiB
    __bf16* Wb = (__bf16*)(ws + (size_t)24 * 1024 * 1024);    // 96 KiB

    wswizzle_kernel<<<12, 256, 0, stream>>>(Wq, Wk, Wv, Wb);
    qkv_kernel<<<N_ * LT_, 32, 0, stream>>>(x, bq, bk, bv, Wb, Qa, Kb, Vb);
    attn_kernel<<<LT_, 256, 0, stream>>>(x, Qa, Kb, Vb, y);
}
